// GraphAttentionLayer_85564338471068
// MI455X (gfx1250) — matrix operand order repaired and checked by reading, not compiled, not run on hardware
//
#include <hip/hip_runtime.h>

// ---------------------------------------------------------------------------
// GraphAttentionLayer, algebraically reduced:
//   softmax logits are constant per row  ->  e == 1/N everywhere
//   out[i,:] = mean_rows(x) @ W^T + b    (same vector for every row i)
// Cost model (MI455X): read x (8MB) + write out (4MB) ~ 12MB @ 23.3TB/s ~ 0.55us.
// WMMA (v_wmma_f32_16x16x32_f16) performs the surviving 256->128 projection.
// ---------------------------------------------------------------------------

typedef __attribute__((ext_vector_type(16))) _Float16 v16h;
typedef __attribute__((ext_vector_type(8)))  float    v8f;

#define N_ROWS 8192
#define F_IN   256
#define F_OUT  128
#define PARTS  256                          // partial-reduction blocks
#define ROWS_PER_PART (N_ROWS / PARTS)      // 32

// d_ws layout (floats):
//   [0, PARTS*F_IN)              : per-block partial column sums of x (256 KB)
//   [PARTS*F_IN, +F_OUT)         : final broadcast row (mean_h + bias)

// ---- Pass 1: partial column sums of x (coalesced, fixed partition) --------
__global__ void gat_colsum_partial(const float* __restrict__ x,
                                   float* __restrict__ part) {
    const int c = threadIdx.x;                       // column 0..255
    const int b = blockIdx.x;                        // 0..PARTS-1
    const float* xr = x + (size_t)b * ROWS_PER_PART * F_IN + c;
    float acc = 0.0f;
#pragma unroll 8
    for (int r = 0; r < ROWS_PER_PART; ++r)
        acc += xr[(size_t)r * F_IN];                 // 256 threads = 1KB coalesced/row
    part[b * F_IN + c] = acc;
}

// ---- Pass 2: reduce partials -> xmean; WMMA 16x256 @ 256x128 -------------
// One workgroup, 256 threads = 8 wave32. Wave w owns output columns [16w,16w+16).
// A-matrix rows are all replicated with xmean, so every D row equals the
// desired result row (robust to M-row placement).
__global__ void gat_rowvec_wmma(const float* __restrict__ part,
                                const float* __restrict__ W,      // [F_OUT][F_IN] row-major
                                const float* __restrict__ bias,   // [F_OUT]
                                float* __restrict__ rowOut) {     // [F_OUT]
    __shared__ _Float16 xm_h[F_IN];

    const int t = threadIdx.x;
    float s = 0.0f;
    for (int p = 0; p < PARTS; ++p)                  // fixed order -> deterministic
        s += part[p * F_IN + t];
    xm_h[t] = (_Float16)(s * (1.0f / (float)N_ROWS));
    __syncthreads();

    const int lane = t & 31;
    const int wave = t >> 5;
    const int n0   = wave * 16;                      // output column tile base
    const int nl   = lane & 15;
    const int half = lane >> 4;

    const float* wrow = W + (size_t)(n0 + nl) * F_IN;

    v8f c = {};
    for (int kk = 0; kk < F_IN; kk += 32) {          // 8 K-chunks of 32
        v16h a, bf;
#pragma unroll
        for (int p = 0; p < 8; ++p) {
            // A (16-bit 16x32 layout, ISA 7.12.2): VGPR p, lanes<16: K = g*16+2q,
            // lanes>=16: +8, where g=p/4, q=p%4. M is the lane%16 row (replicated).
            const int abase = ((p >> 2) * 16) + ((p & 3) * 2) + (half ? 8 : 0);
            a[2 * p]     = xm_h[kk + abase];
            a[2 * p + 1] = xm_h[kk + abase + 1];
            // B = W^T (32x16): VGPR p, lane n: rows K = same as A (abase, +1), col n%16.
            const int kb = kk + abase;
            bf[2 * p]     = (_Float16)wrow[kb];
            bf[2 * p + 1] = (_Float16)wrow[kb + 1];
        }
        // (neg_a, A, neg_b, B, c_mod, C, reuse_a, reuse_b)
        c = __builtin_amdgcn_wmma_f32_16x16x32_f16(false, a, false, bf,
                                                   (short)0, c, false, false);
    }

    // D row 0 lives in c[0] of lanes 0..15 (all rows are identical anyway).
    if (lane < 16)
        rowOut[n0 + lane] = c[0] + bias[n0 + lane];
}

// ---- Pass 3: broadcast the 128-float row to all 8192 output rows ----------
__global__ void gat_broadcast(const float* __restrict__ rowOut,
                              float4* __restrict__ out4) {
    __shared__ float4 srow[F_OUT / 4];               // 32 x float4
    if (threadIdx.x < F_OUT / 4)
        srow[threadIdx.x] = ((const float4*)rowOut)[threadIdx.x];
    __syncthreads();

    const int total = N_ROWS * (F_OUT / 4);          // 262144 float4 stores
    for (int i = blockIdx.x * blockDim.x + threadIdx.x; i < total;
         i += gridDim.x * blockDim.x)
        out4[i] = srow[i & (F_OUT / 4 - 1)];         // global_store_b128, coalesced
}

extern "C" void kernel_launch(void* const* d_in, const int* in_sizes, int n_in,
                              void* d_out, int out_size, void* d_ws, size_t ws_size,
                              hipStream_t stream) {
    (void)in_sizes; (void)n_in; (void)out_size; (void)ws_size;
    // setup_inputs order: x, adj, W, b, Wa, ba  (adj/Wa/ba algebraically unused)
    const float* x    = (const float*)d_in[0];
    const float* W    = (const float*)d_in[2];
    const float* bias = (const float*)d_in[3];
    float* out = (float*)d_out;

    float* part   = (float*)d_ws;                    // PARTS*F_IN floats
    float* rowOut = part + PARTS * F_IN;             // F_OUT floats

    gat_colsum_partial<<<PARTS, F_IN, 0, stream>>>(x, part);
    gat_rowvec_wmma<<<1, 256, 0, stream>>>(part, W, bias, rowOut);
    gat_broadcast<<<256, 256, 0, stream>>>(rowOut, (float4*)out);
}
